// SpectralAttentionLayer_81913616269394
// MI455X (gfx1250) — compile-verified, matrix operand order checked by reading, not run on hardware
//
#include <hip/hip_runtime.h>
#include <math.h>

// ---------------- problem constants ----------------
#define BB 4
#define SS 4096
#define DD 1024
#define NH 16
#define HD 64
#define NTOT 4194304          // B-flattened signal length S*D
#define KMIN 41944            // freqs > 0.01  -> k >= ceil(0.01*NTOT)
#define KMAX 2097151          // freqs < 0.5   -> k <= NTOT/2 - 1
#define MASK_CNT 2055208.0

typedef __attribute__((ext_vector_type(16))) _Float16 v16h;
typedef __attribute__((ext_vector_type(8)))  _Float16 v8h;
typedef __attribute__((ext_vector_type(8)))  float    v8f;

__device__ __forceinline__ float2 cmulf(float2 a, float2 b) {
    return make_float2(a.x * b.x - a.y * b.y, a.x * b.y + a.y * b.x);
}

// Iterative radix-2 DIT FFT over LDS (input must be bit-reversed). 256 threads.
__device__ __forceinline__ void fft_lds(float2* z, int log2n, int t) {
    const int nbf = 1 << (log2n - 1);
    for (int s = 0; s < log2n; ++s) {
        const int half = 1 << s;
        for (int i = t; i < nbf; i += 256) {
            const int pos = i & (half - 1);
            const int i0  = ((i >> s) << (s + 1)) + pos;
            const int i1  = i0 + half;
            float sn, cs;
            __sincosf(-3.14159265358979f * (float)pos / (float)half, &sn, &cs);
            float2 a = z[i0], b = z[i1];
            float2 tw = make_float2(cs * b.x - sn * b.y, cs * b.y + sn * b.x);
            z[i0] = make_float2(a.x + tw.x, a.y + tw.y);
            z[i1] = make_float2(a.x - tw.x, a.y - tw.y);
        }
        __syncthreads();
    }
}

// ---------------- stage A: masked log-k stats (constants of n) ----------------
__global__ __launch_bounds__(256) void k_stats_pass(double* __restrict__ dpart) {
    const double inv_n = 1.0 / (double)NTOT;
    const int blk = blockIdx.x, t = threadIdx.x;
    const long k0 = (long)KMIN + (long)blk * 1004;
    double s1 = 0.0, s2 = 0.0;
    for (int i = t; i < 1004; i += 256) {
        long k = k0 + i;
        if (k <= (long)KMAX) {
            double l = log((double)k * inv_n + 1e-10);
            s1 += l; s2 += l * l;
        }
    }
    __shared__ double r1[256], r2[256];
    r1[t] = s1; r2[t] = s2; __syncthreads();
    for (int o = 128; o > 0; o >>= 1) {
        if (t < o) { r1[t] += r1[t + o]; r2[t] += r2[t + o]; }
        __syncthreads();
    }
    if (t == 0) { dpart[blk] = r1[0]; dpart[2048 + blk] = r2[0]; }
}

__global__ __launch_bounds__(256) void k_stats_final(const double* __restrict__ dpart,
                                                     float* __restrict__ stats) {
    __shared__ double r1[256], r2[256];
    int t = threadIdx.x;
    double a = 0.0, b = 0.0;
    for (int i = t; i < 2048; i += 256) { a += dpart[i]; b += dpart[2048 + i]; }
    r1[t] = a; r2[t] = b; __syncthreads();
    for (int o = 128; o > 0; o >>= 1) {
        if (t < o) { r1[t] += r1[t + o]; r2[t] += r2[t + o]; }
        __syncthreads();
    }
    if (t == 0) {
        double mu  = r1[0] / MASK_CNT;
        double var = r2[0] - MASK_CNT * mu * mu;   // sum of centered squares
        stats[0] = (float)mu; stats[1] = (float)var;
    }
}

// ---------------- stage A: four-step 4M FFT (2048 x 2048), per batch ----------------
// pass1: column FFTs over j1 (stride N2=2048), twiddle by e^{-2pi i k1 j2 / n}, 4 cols/block.
__global__ __launch_bounds__(256) void k_fft_pass1(const float* __restrict__ x,
                                                   float2* __restrict__ aw, int b) {
    __shared__ float2 z[4][2048];                 // 64 KB
    const int t = threadIdx.x, c0 = blockIdx.x * 4;
    const float* xb = x + ((size_t)b << 22);
    for (int i = t; i < 8192; i += 256) {
        int j1 = i >> 2, c = i & 3;
        float v = xb[(size_t)j1 * 2048 + c0 + c];
        z[c][__brev((unsigned)j1) >> 21] = make_float2(v, 0.f);
    }
    __syncthreads();
    for (int s = 0; s < 11; ++s) {
        const int half = 1 << s;
        for (int i = t; i < 4096; i += 256) {
            int c = i >> 10, j = i & 1023;
            int pos = j & (half - 1);
            int i0 = ((j >> s) << (s + 1)) + pos, i1 = i0 + half;
            float sn, cs;
            __sincosf(-3.14159265358979f * (float)pos / (float)half, &sn, &cs);
            float2 a = z[c][i0], bb = z[c][i1];
            float2 tw = make_float2(cs * bb.x - sn * bb.y, cs * bb.y + sn * bb.x);
            z[c][i0] = make_float2(a.x + tw.x, a.y + tw.y);
            z[c][i1] = make_float2(a.x - tw.x, a.y - tw.y);
        }
        __syncthreads();
    }
    for (int i = t; i < 8192; i += 256) {
        int k1 = i >> 2, c = i & 3, j2 = c0 + c;
        float fr = (float)(k1 * j2) * (1.0f / (float)NTOT);   // exact: < 2^23
        float sn, cs;
        __sincosf(-6.28318530717958f * fr, &sn, &cs);
        float2 v = cmulf(z[c][k1], make_float2(cs, sn));
        aw[(size_t)k1 * 2048 + j2] = v;
    }
}

// pass2: row FFT over j2 (contiguous), fuse slope-regression partial sums.
__global__ __launch_bounds__(256) void k_fft_pass2(const float2* __restrict__ aw,
                                                   const float* __restrict__ stats,
                                                   float* __restrict__ partial, int b) {
    __shared__ float2 z[2048];
    __shared__ float red[256];
    const int t = threadIdx.x, k1 = blockIdx.x;
    for (int i = t; i < 2048; i += 256)
        z[__brev((unsigned)i) >> 21] = aw[(size_t)k1 * 2048 + i];
    __syncthreads();
    fft_lds(z, 11, t);
    const float mu = stats[0];
    float acc = 0.f;
    for (int k2 = t; k2 < 2048; k2 += 256) {
        long k = (long)k2 * 2048 + k1;            // X[k2*N1 + k1]
        if (k >= (long)KMIN && k <= (long)KMAX) {
            float2 v = z[k2];
            float P  = v.x * v.x + v.y * v.y;
            float lk = logf((float)k * (1.0f / (float)NTOT) + 1e-10f) - mu;
            acc += lk * logf(P + 1e-10f);
        }
    }
    red[t] = acc; __syncthreads();
    for (int o = 128; o > 0; o >>= 1) {
        if (t < o) red[t] += red[t + o];
        __syncthreads();
    }
    if (t == 0) partial[b * 2048 + k1] = red[0];
}

__global__ __launch_bounds__(256) void k_alpha(const float* __restrict__ partial,
                                               const float* __restrict__ stats,
                                               float* __restrict__ alpha) {
    __shared__ double red[256];
    int t = threadIdx.x;
    for (int b = 0; b < BB; ++b) {
        double a = 0.0;
        for (int i = t; i < 2048; i += 256) a += (double)partial[b * 2048 + i];
        red[t] = a; __syncthreads();
        for (int o = 128; o > 0; o >>= 1) { if (t < o) red[t] += red[t + o]; __syncthreads(); }
        if (t == 0) {
            double cov  = red[0];
            double var  = (double)stats[1];
            double beta = -cov / (var + 1e-10);
            double Dim  = fmin(fmax((3.0 - beta) * 0.5, 0.5), 1.5);
            double al   = fmin(fmax(1.0 + 0.8 * (Dim - 1.0), 0.1), 3.0);
            alpha[b] = (float)al;
        }
        __syncthreads();
    }
}

// ---------------- build real symmetric filter H[b][h][0..2048] ----------------
__global__ __launch_bounds__(256) void k_buildH(const float* __restrict__ alpha,
                                                const float* __restrict__ ab,
                                                const float* __restrict__ ps,
                                                float* __restrict__ H) {
    __shared__ float lk[SS];
    __shared__ float red[256];
    __shared__ float s_mu, s_istd;
    const int t = threadIdx.x;
    for (int s = t; s < SS; s += 256) {
        int m = (s <= SS / 2) ? s : (SS - s);     // |fftfreq| * S
        lk[s] = logf((float)m * (1.0f / (float)SS) + 1e-8f);
    }
    __syncthreads();
    float a = 0.f;
    for (int s = t; s < SS; s += 256) a += lk[s];
    red[t] = a; __syncthreads();
    for (int o = 128; o > 0; o >>= 1) { if (t < o) red[t] += red[t + o]; __syncthreads(); }
    if (t == 0) s_mu = red[0] * (1.0f / (float)SS);
    __syncthreads();
    const float mu = s_mu;
    float q = 0.f;
    for (int s = t; s < SS; s += 256) { float d = lk[s] - mu; q += d * d; }
    red[t] = q; __syncthreads();
    for (int o = 128; o > 0; o >>= 1) { if (t < o) red[t] += red[t + o]; __syncthreads(); }
    if (t == 0) s_istd = 1.0f / (sqrtf(red[0] / (float)(SS - 1)) + 1e-8f);  // unbiased std
    __syncthreads();
    const float istd = s_istd;
    for (int idx = t; idx < BB * NH * 2049; idx += 256) {
        int f = idx % 2049, h = (idx / 2049) % NH, b = idx / (2049 * NH);
        float xn = (lk[f] - mu) * istd;
        float g  = 0.5f * xn * (1.0f + erff(xn * 0.70710678118f));   // exact GELU
        H[idx] = cosf(ab[h] * alpha[b] * g + ps[h]);
    }
}

// ---------------- per-head filtering: z = x[ch] + i*x[ch+1], y = ifft(fft(z)*H) ----------------
__global__ __launch_bounds__(256) void k_filter(const float* __restrict__ x,
                                                const float* __restrict__ H,
                                                _Float16* __restrict__ xf) {
    __shared__ float2 z[SS];                      // 32 KB
    const int t = threadIdx.x, blk = blockIdx.x;  // blk = b*512 + h*32 + p
    const int b = blk >> 9, h = (blk >> 5) & 15, p = blk & 31;
    const int ch = h * HD + 2 * p;
    const float* xb = x + (size_t)b * SS * DD + ch;
    for (int s = t; s < SS; s += 256) {
        float2 v = *(const float2*)(xb + (size_t)s * DD);
        z[__brev((unsigned)s) >> 20] = v;
    }
    __syncthreads();
    fft_lds(z, 12, t);                            // forward FFT
    const float* Hb = H + ((size_t)b * NH + h) * 2049;
    for (int f = t; f < SS; f += 256) {
        int fm = (f <= 2048) ? f : (SS - f);
        float hv = Hb[fm];
        float2 v = z[f];
        z[f] = make_float2(v.x * hv, -v.y * hv);  // multiply by real H, conjugate for inverse
    }
    __syncthreads();
    for (int s = t; s < SS; s += 256) {           // bit-reverse permute in place
        int r = __brev((unsigned)s) >> 20;
        if (s < r) { float2 tmp = z[s]; z[s] = z[r]; z[r] = tmp; }
    }
    __syncthreads();
    fft_lds(z, 12, t);                            // ifft(W) = conj(fft(conj(W)))/N
    _Float16* o1 = xf + ((size_t)b * DD + ch) * SS;  // [b][d][s] layout for GEMM
    _Float16* o2 = o1 + SS;
    const float inv = 1.0f / (float)SS;
    for (int s = t; s < SS; s += 256) {
        float2 v = z[s];
        o1[s] = (_Float16)(v.x * inv);            // Re -> channel ch
        o2[s] = (_Float16)(-v.y * inv);           // Im -> channel ch+1
    }
}

__global__ __launch_bounds__(256) void k_convW(const float* __restrict__ W,
                                               _Float16* __restrict__ Wh) {
    for (size_t i = (size_t)blockIdx.x * 256 + threadIdx.x; i < (size_t)DD * DD;
         i += (size_t)gridDim.x * 256)
        Wh[i] = (_Float16)W[i];
}

// ---------------- WMMA GEMM: out = xf @ W^T + b + x (fused residual) ----------------
// Block: 256 thr = 8 wave32 waves. Block tile: M=128 (16/wave), N=128 (8 N-tiles/wave).
// 8 v_wmma per wave per k-step between barriers; A re-streamed only 8x total.
__global__ __launch_bounds__(256) void k_gemm(const _Float16* __restrict__ xf,
                                              const _Float16* __restrict__ Wh,
                                              const float* __restrict__ bias,
                                              const float* __restrict__ x,
                                              float* __restrict__ out) {
    __shared__ _Float16 Al[128 * 48];             // A tile [m][k], row stride 48 (16B-aligned)
    __shared__ _Float16 Bl[128 * 48];             // B tile [n][k]
    const int t = threadIdx.x;
    const int w = t >> 5, L = t & 31;             // wave32
    const int ln = L & 15, g = L >> 4, g8 = g * 8;
    const int m0 = blockIdx.x * 128, n0 = blockIdx.y * 128;
    const int bi = m0 >> 12, s0 = m0 & (SS - 1);  // M strip lies within one batch (4096 % 128 == 0)
    const _Float16* xfb = xf + (size_t)bi * DD * SS;

    v8f acc[8] = {};
    for (int kt = 0; kt < 32; ++kt) {
        const int k0 = kt * 32;
        // stage A: 32 k-rows x 128 m, contiguous in m ([b][d][s] layout)
        for (int i = t; i < 512; i += 256) {
            int k = i >> 4, mo = (i & 15) * 8;
            union { float4 f4; _Float16 hh[8]; } u;
            u.f4 = *(const float4*)(xfb + (size_t)(k0 + k) * SS + s0 + mo);
#pragma unroll
            for (int j = 0; j < 8; ++j) Al[(mo + j) * 48 + k] = u.hh[j];
        }
        // stage B: 128 n-rows x 32 k, contiguous in k (W row-major)
        for (int i = t; i < 512; i += 256) {
            int n = i >> 2, ko = (i & 3) * 8;
            union { float4 f4; v8h h8; } u;
            u.f4 = *(const float4*)(Wh + (size_t)(n0 + n) * DD + k0 + ko);
            *(v8h*)&Bl[n * 48 + ko] = u.h8;
        }
        if (kt < 31) {
            __builtin_prefetch(xfb + (size_t)(k0 + 32) * SS + s0 + (t & 15) * 8, 0, 1);
            __builtin_prefetch(Wh + (size_t)(n0 + (t >> 1)) * DD + k0 + 32, 0, 1);
        }
        __syncthreads();

        // A fragment: lane L -> M = L%16; v16h = K {g8..g8+7, 16+g8..16+g8+7}
        v8h alo = *(const v8h*)&Al[(w * 16 + ln) * 48 + g8];
        v8h ahi = *(const v8h*)&Al[(w * 16 + ln) * 48 + 16 + g8];
        v16h a;
#pragma unroll
        for (int j = 0; j < 8; ++j) { a[j] = alo[j]; a[8 + j] = ahi[j]; }
#pragma unroll
        for (int nt = 0; nt < 8; ++nt) {
            v8h blo = *(const v8h*)&Bl[(nt * 16 + ln) * 48 + g8];
            v8h bhi = *(const v8h*)&Bl[(nt * 16 + ln) * 48 + 16 + g8];
            v16h bb;
#pragma unroll
            for (int j = 0; j < 8; ++j) { bb[j] = blo[j]; bb[8 + j] = bhi[j]; }
            acc[nt] = __builtin_amdgcn_wmma_f32_16x16x32_f16(
                false, a, false, bb, (short)0, acc[nt], false, false);
        }
        __syncthreads();
    }
    // epilogue: C layout lane L, vgpr r -> M = r + 8*(L/16), N = L%16
#pragma unroll
    for (int nt = 0; nt < 8; ++nt) {
        const int gn = n0 + nt * 16 + ln;
        const float bv = bias[gn];
#pragma unroll
        for (int r = 0; r < 8; ++r) {
            const int gm = m0 + w * 16 + r + 8 * g;
            const size_t off = (size_t)gm * DD + gn;
            out[off] = acc[nt][r] + bv + x[off];  // residual fused
        }
    }
}

// ---------------- in-place LayerNorm over last dim (biased var) ----------------
__global__ __launch_bounds__(256) void k_ln(float* __restrict__ out,
                                            const float* __restrict__ gamma,
                                            const float* __restrict__ beta) {
    __shared__ float red[256];
    __shared__ float s_mu, s_rstd;
    const int t = threadIdx.x;
    const size_t row = (size_t)blockIdx.x * DD;
    float v[4];
#pragma unroll
    for (int j = 0; j < 4; ++j) v[j] = out[row + t + 256 * j];
    float s = v[0] + v[1] + v[2] + v[3];
    red[t] = s; __syncthreads();
    for (int o = 128; o > 0; o >>= 1) { if (t < o) red[t] += red[t + o]; __syncthreads(); }
    if (t == 0) s_mu = red[0] * (1.0f / (float)DD);
    __syncthreads();
    const float mu = s_mu;
    float q = 0.f;
#pragma unroll
    for (int j = 0; j < 4; ++j) { float d = v[j] - mu; q += d * d; }
    red[t] = q; __syncthreads();
    for (int o = 128; o > 0; o >>= 1) { if (t < o) red[t] += red[t + o]; __syncthreads(); }
    if (t == 0) s_rstd = rsqrtf(red[0] * (1.0f / (float)DD) + 1e-5f);
    __syncthreads();
    const float rstd = s_rstd;
#pragma unroll
    for (int j = 0; j < 4; ++j) {
        int c = t + 256 * j;
        out[row + c] = (v[j] - mu) * rstd * gamma[c] + beta[c];
    }
}

// ---------------- launch ----------------
extern "C" void kernel_launch(void* const* d_in, const int* in_sizes, int n_in,
                              void* d_out, int out_size, void* d_ws, size_t ws_size,
                              hipStream_t stream) {
    (void)in_sizes; (void)n_in; (void)out_size; (void)ws_size;
    const float* x     = (const float*)d_in[0];
    const float* ab    = (const float*)d_in[1];
    const float* ps    = (const float*)d_in[2];
    const float* Wout  = (const float*)d_in[3];
    const float* bout  = (const float*)d_in[4];
    const float* gamma = (const float*)d_in[5];
    const float* beta  = (const float*)d_in[6];
    float* out = (float*)d_out;

    char* w = (char*)d_ws;
    size_t off = 0;
    auto take = [&](size_t bytes) { size_t o = off; off = (off + bytes + 255) & ~(size_t)255; return o; };
    float2*   aw      = (float2*)(w + take((size_t)NTOT * sizeof(float2)));        // 32 MB (reused per batch)
    double*   dpart   = (double*)(w + take(4096 * sizeof(double)));
    float*    stats   = (float*)(w + take(16 * sizeof(float)));
    float*    alphaB  = (float*)(w + take(16 * sizeof(float)));
    float*    partial = (float*)(w + take((size_t)BB * 2048 * sizeof(float)));
    float*    Hbuf    = (float*)(w + take((size_t)BB * NH * 2049 * sizeof(float)));
    _Float16* xf      = (_Float16*)(w + take((size_t)BB * DD * SS * sizeof(_Float16)));  // 32 MB
    _Float16* Wh      = (_Float16*)(w + take((size_t)DD * DD * sizeof(_Float16)));       // 2 MB

    k_stats_pass<<<2048, 256, 0, stream>>>(dpart);
    k_stats_final<<<1, 256, 0, stream>>>(dpart, stats);
    for (int b = 0; b < BB; ++b) {
        k_fft_pass1<<<512, 256, 0, stream>>>(x, aw, b);
        k_fft_pass2<<<2048, 256, 0, stream>>>(aw, stats, partial, b);
    }
    k_alpha<<<1, 256, 0, stream>>>(partial, stats, alphaB);
    k_buildH<<<1, 256, 0, stream>>>(alphaB, ab, ps, Hbuf);
    k_convW<<<1024, 256, 0, stream>>>(Wout, Wh);
    k_filter<<<BB * NH * 32, 256, 0, stream>>>(x, Hbuf, xf);
    k_gemm<<<dim3(128, 8), 256, 0, stream>>>(xf, Wh, bout, x, out);
    k_ln<<<BB * SS, 256, 0, stream>>>(out, gamma, beta);
}
